// SENT_RET_41549513622441
// MI455X (gfx1250) — matrix operand order repaired and checked by reading, not compiled, not run on hardware
//
#include <hip/hip_runtime.h>
#include <math.h>

// ---------------------------------------------------------------------------
// MI455X (gfx1250) implementation.
// Heavy work (doc conv blocks + similarity GEMMs) runs on bf16 WMMA
// (v_wmma_f32_16x16x32_bf16, f32 accumulate), wave32; A and B share contiguous k order (k = 16*half + s).
// LDS holds sentences with zero guard rows so the conv k-shift is branch-free.
// ---------------------------------------------------------------------------

typedef __bf16 bf16_t;
typedef __attribute__((ext_vector_type(8)))  bf16_t v8bf;
typedef __attribute__((ext_vector_type(16))) bf16_t v16bf;
typedef __attribute__((ext_vector_type(8)))  float  v8f;

#define DR      300      // real feature dim
#define DP      320      // padded to multiple of 32
#define SL      64       // tokens per sentence
#define NQ      32       // question tokens
#define TS      2        // sentences per block
#define MROWS   (TS*SL)  // 128 logical rows
#define PROWS   (TS*(SL+2))  // 132 physical rows (guard row before/after each sentence)
#define NT_CONV 20       // DP/16
#define KT      10       // DP/32
#define NWAVES  8
#define EPS_OH  0.999f

// physical LDS row for logical row r (sentence = r>>6): +1 guard per sentence edge
__device__ __forceinline__ int phys_row(int r) { return r + 1 + ((r >> 6) << 1); }

// ---- workspace layout (bytes) ----------------------------------------------
#define WPACK_FRAGS   (2*3*KT*NT_CONV)            // 1200
#define WPACK_ELEMS   (WPACK_FRAGS*512)           // 614400 bf16
#define WS_WPACK      0
#define WS_QN         (WS_WPACK + WPACK_ELEMS*2)  // 2*KT fragments * 512 bf16
#define WS_QCN        (WS_QN  + 2*KT*512*2)
#define WS_QW         (WS_QCN + 2*KT*512*2)       // 32 f32
#define WS_Q1         (WS_QW  + 32*4)             // 32*300 f32
#define WS_QCTX       (WS_Q1  + NQ*DR*4)          // 32*300 f32

// ---------------------------------------------------------------------------
__device__ __forceinline__ v16bf pack16(v8bf a, v8bf b) {
    v16bf r;
#pragma unroll
    for (int i = 0; i < 8; ++i) { r[i] = a[i]; r[i + 8] = b[i]; }
    return r;
}

// A fragment (16x32 bf16, wave32): lane holds row M=lane&15; lanes 0-15 take
// K [0,16), lanes 16-31 take K [16,32).
__device__ __forceinline__ v16bf load_a_frag(const bf16_t* p) {
    v8bf c0 = *(const v8bf*)(p);
    v8bf c1 = *(const v8bf*)(p + 8);
    return pack16(c0, c1);
}

__device__ __forceinline__ v8f wmma_bf16(v16bf a, v16bf b, v8f c) {
    return __builtin_amdgcn_wmma_f32_16x16x32_bf16(
        /*neg_a=*/false, a, /*neg_b=*/false, b,
        /*c_mod=*/(short)0, c, /*reuse_a=*/false, /*reuse_b=*/false);
}

// ---------------------------------------------------------------------------
// conv block as GEMM:  y[l,o] = lrelu( sum_{k,i} x[l+k-1,i]*W[o,i,k] + b[o] ) + x[l,o]
// Guard rows in LDS make the k-shift unconditional.  Each wave owns N-tiles
// (stride NWAVES) and keeps 8 M-tile accumulators so every B fragment load
// (from the L2-resident pre-swizzled weight pack) is reused 8x.
// ---------------------------------------------------------------------------
__device__ __forceinline__ void conv_block_gemm(const bf16_t* Xin, bf16_t* Xout,
                                                const bf16_t* __restrict__ wpack,
                                                const float* __restrict__ bias,
                                                int wave, int lane) {
    const int half = lane >> 4;
    const int mrow = lane & 15;
    const int ncol = lane & 15;

    // per-lane physical row base pointer for each M-tile (at this lane's K half)
    const bf16_t* rowp[8];
#pragma unroll
    for (int mt = 0; mt < 8; ++mt)
        rowp[mt] = Xin + (size_t)phys_row(mt * 16 + mrow) * DP + half * 16;

    for (int nt = wave; nt < NT_CONV; nt += NWAVES) {
        v8f zf = {};
        v8f acc[8];
#pragma unroll
        for (int mt = 0; mt < 8; ++mt) acc[mt] = zf;

        for (int kt = 0; kt < KT; ++kt) {
#pragma unroll
            for (int k = 0; k < 3; ++k) {
                const bf16_t* bp =
                    wpack + (size_t)((k * KT + kt) * NT_CONV + nt) * 512 + lane * 16;
                v16bf bfrag = *(const v16bf*)bp;
                const int soff = (k - 1) * DP;   // row shift, always in-bounds (guards)
#pragma unroll
                for (int mt = 0; mt < 8; ++mt) {
                    v16bf afrag = load_a_frag(rowp[mt] + soff + kt * 32);
                    acc[mt] = wmma_bf16(afrag, bfrag, acc[mt]);
                }
            }
        }

        // epilogue: + bias, leaky relu(0.1), + residual, store bf16
        const int col = nt * 16 + ncol;
        const float bv = (col < DR) ? bias[col] : 0.0f;
#pragma unroll
        for (int mt = 0; mt < 8; ++mt) {
#pragma unroll
            for (int v = 0; v < 8; ++v) {
                const int row = mt * 16 + v + half * 8;
                const int pr  = phys_row(row);
                float y = acc[mt][v] + bv;
                y = (y > 0.0f) ? y : 0.1f * y;
                y += (float)Xin[(size_t)pr * DP + col];
                Xout[(size_t)pr * DP + col] = (bf16_t)y;
            }
        }
    }
}

// ---------------------------------------------------------------------------
// similarity GEMM: S[l,q] = (1/||x_l||) * sum_i X[l,i] * qn[q,i]
// M=128 (one M-tile per wave), N=32 (2 tiles), K=DP
// ---------------------------------------------------------------------------
__device__ __forceinline__ void sim_gemm(const bf16_t* Xin,
                                         const bf16_t* __restrict__ qpack,
                                         float* Sout, const float* RN,
                                         int wave, int lane) {
    const int half = lane >> 4;
    const int mrow = lane & 15;
    const int ncol = lane & 15;
    const int mt = wave;   // 8 waves cover the 8 M-tiles
    const bf16_t* arow = Xin + (size_t)phys_row(mt * 16 + mrow) * DP + half * 16;

    for (int nt = 0; nt < 2; ++nt) {
        v8f acc = {};
        for (int kt = 0; kt < KT; ++kt) {
            const bf16_t* bp = qpack + (size_t)(kt * 2 + nt) * 512 + lane * 16;
            v16bf bfrag = *(const v16bf*)bp;
            v16bf afrag = load_a_frag(arow + kt * 32);
            acc = wmma_bf16(afrag, bfrag, acc);
        }
        const int colq = nt * 16 + ncol;
#pragma unroll
        for (int v = 0; v < 8; ++v) {
            const int row = mt * 16 + v + half * 8;
            Sout[row * NQ + colq] = acc[v] * RN[row];
        }
    }
}

// ---------------------------------------------------------------------------
// Prep kernel 1: pack conv weights (f32, [O][I][3]) into bf16 B fragments.
// ---------------------------------------------------------------------------
__global__ void pack_w_kernel(const float* __restrict__ w1,
                              const float* __restrict__ w2,
                              bf16_t* __restrict__ wpack) {
    int g = blockIdx.x * 256 + threadIdx.x;      // < 38400 (== 1200 frags * 32)
    int lane = g & 31;
    int frag = g >> 5;
    int nt = frag % NT_CONV;
    int kt = (frag / NT_CONV) % KT;
    int k  = (frag / (NT_CONV * KT)) % 3;
    int c  =  frag / (NT_CONV * KT * 3);
    const float* W = c ? w2 : w1;
    bf16_t* dst = wpack + (size_t)frag * 512 + lane * 16;
    int o  = nt * 16 + (lane & 15);
    int ib = kt * 32 + (lane >> 4) * 16;
#pragma unroll
    for (int j = 0; j < 16; ++j) {
        int i = ib + j;
        float v = (i < DR && o < DR) ? W[((size_t)o * DR + i) * 3 + k] : 0.0f;
        dst[j] = (bf16_t)v;
    }
}

// ---------------------------------------------------------------------------
// Prep kernel 2: one conv block on the question (tiny; f32 scalar math).
// ---------------------------------------------------------------------------
__global__ void qconv_kernel(const float* __restrict__ x,
                             const float* __restrict__ W,
                             const float* __restrict__ b,
                             float* __restrict__ y) {
    int g = blockIdx.x * 256 + threadIdx.x;
    if (g >= NQ * DR) return;
    int l = g / DR, o = g % DR;
    float s = 0.0f;
#pragma unroll
    for (int k = 0; k < 3; ++k) {
        int sr = l + k - 1;
        if (sr < 0 || sr >= NQ) continue;
        const float* xr = x + (size_t)sr * DR;
        const float* wr = W + (size_t)o * DR * 3 + k;
        for (int i = 0; i < DR; ++i) s += xr[i] * wr[i * 3];
    }
    s += b[o];
    s = (s > 0.0f) ? s : 0.1f * s;
    y[g] = s + x[(size_t)l * DR + o];
}

// ---------------------------------------------------------------------------
// Prep kernel 3: q_weights softmax + normalized qn/qcn packed as B fragments.
// ---------------------------------------------------------------------------
__global__ __launch_bounds__(256)
void qfinal_kernel(const float* __restrict__ qemb,
                   const float* __restrict__ qctx,
                   const float* __restrict__ qidf,
                   const float* __restrict__ qww,
                   const float* __restrict__ qwb,
                   bf16_t* __restrict__ qnpack,
                   bf16_t* __restrict__ qcnpack,
                   float* __restrict__ qwout) {
    __shared__ float lg[NQ];
    __shared__ float nrm[2 * NQ];
    int t = threadIdx.x;

    if (t < NQ) {
        float z = qwb[0] + qidf[t] * qww[DR];
        const float* xr = qctx + (size_t)t * DR;
        for (int i = 0; i < DR; ++i) z += xr[i] * qww[i];
        lg[t] = z;
    }
    if (t >= NQ && t < 3 * NQ) {   // 64 threads: row norms of qemb / qctx
        int u = t - NQ; int mat = u >> 5; int q = u & 31;
        const float* xr = (mat ? qctx : qemb) + (size_t)q * DR;
        float ss = 0.0f;
        for (int i = 0; i < DR; ++i) ss += xr[i] * xr[i];
        nrm[u] = rsqrtf(ss + 1e-20f);
    }
    __syncthreads();

    if (t == 0) {
        float mx = lg[0];
        for (int q = 1; q < NQ; ++q) mx = fmaxf(mx, lg[q]);
        float se = 0.0f;
        for (int q = 0; q < NQ; ++q) se += expf(lg[q] - mx);
        float inv = 1.0f / se;
        for (int q = 0; q < NQ; ++q) qwout[q] = expf(lg[q] - mx) * inv;
    }

    // pack qn / qcn into B fragments (frag index = kt*2+nt)
    for (int slot = t; slot < 2 * 2 * KT * 32; slot += 256) {
        int lane = slot & 31;
        int rem  = slot >> 5;        // 0..39
        int mat  = rem / (2 * KT);
        int fi   = rem % (2 * KT);   // == kt*2+nt
        int nt = fi & 1;
        int kt = fi >> 1;
        const float* src = mat ? qctx : qemb;
        bf16_t* dst = (mat ? qcnpack : qnpack) + (size_t)fi * 512 + lane * 16;
        int qc = nt * 16 + (lane & 15);
        int ib = kt * 32 + (lane >> 4) * 16;
        float rn = nrm[mat * NQ + qc];
#pragma unroll
        for (int j = 0; j < 16; ++j) {
            int i = ib + j;
            float v = (i < DR) ? src[(size_t)qc * DR + i] * rn : 0.0f;
            dst[j] = (bf16_t)v;
        }
    }
}

// ---------------------------------------------------------------------------
// Main kernel: one block handles TS=2 sentences.
// LDS (dynamic, ~204KB): XA,XB bf16 132x320 (guard rows) | S1,S2 f32 128x32 |
// norms, partials, small buffers.
// ---------------------------------------------------------------------------
__global__ __launch_bounds__(256)
void sent_main_kernel(const float* __restrict__ doc,
                      const float* __restrict__ gaf,
                      const bf16_t* __restrict__ wpack1,
                      const bf16_t* __restrict__ wpack2,
                      const float* __restrict__ b1,
                      const float* __restrict__ b2,
                      const bf16_t* __restrict__ qnpack,
                      const bf16_t* __restrict__ qcnpack,
                      const float* __restrict__ qw,
                      const float* __restrict__ m1w, const float* __restrict__ m1b,
                      const float* __restrict__ m2w, const float* __restrict__ m2b,
                      const float* __restrict__ outw,
                      float* __restrict__ out) {
    extern __shared__ char smem[];
    bf16_t* XA   = (bf16_t*)smem;                 // x0 -> later sent_ctx
    bf16_t* XB   = XA + (size_t)PROWS * DP;       // x1
    float*  S1   = (float*)(XB + (size_t)PROWS * DP);   // sim_insens [128][32]
    float*  S2   = S1 + MROWS * NQ;               // sim_sens
    float*  RN   = S2 + MROWS * NQ;               // 128 reciprocal norms
    float*  PART = RN + MROWS;                    // 256 partial sumsq
    float*  LO   = PART + 256;                    // 64 mlp outputs
    float*  QWS  = LO + 64;                       // 32 q weights

    const int t    = threadIdx.x;
    const int wave = t >> 5;
    const int lane = t & 31;
    const int blk  = blockIdx.x;

    if (t < NQ) QWS[t] = qw[t];

    // ---- zero the guard rows of XA and XB (once) ----------------------------
    {
        const int guards[4] = {0, SL + 1, SL + 2, 2 * SL + 3};   // 0,65,66,131
        for (int i = t; i < 4 * DP; i += 256) {
            int gr = guards[i / DP], c = i % DP;
            XA[(size_t)gr * DP + c] = (bf16_t)0.0f;
            XB[(size_t)gr * DP + c] = (bf16_t)0.0f;
        }
    }

    // ---- load 2 sentences, convert to bf16, row sum-of-squares -------------
    {
        int r = t >> 1, p = t & 1;
        const int pr = phys_row(r);
        const float* src = doc + ((size_t)(blk * TS) * SL + r) * DR;
        float ss = 0.0f;
        for (int ch = p; ch < 75; ch += 2) {
            float4 vv = *(const float4*)(src + ch * 4);
            ss += vv.x * vv.x + vv.y * vv.y + vv.z * vv.z + vv.w * vv.w;
            XA[(size_t)pr * DP + ch * 4 + 0] = (bf16_t)vv.x;
            XA[(size_t)pr * DP + ch * 4 + 1] = (bf16_t)vv.y;
            XA[(size_t)pr * DP + ch * 4 + 2] = (bf16_t)vv.z;
            XA[(size_t)pr * DP + ch * 4 + 3] = (bf16_t)vv.w;
        }
        for (int c2 = DR + p * 10; c2 < DR + p * 10 + 10; ++c2)
            XA[(size_t)pr * DP + c2] = (bf16_t)0.0f;   // zero the K/N padding
        PART[t] = ss;
    }
    __syncthreads();
    if (t < MROWS) RN[t] = rsqrtf(PART[2 * t] + PART[2 * t + 1] + 1e-20f);
    __syncthreads();

    // ---- sim_insens (raw embeddings), then two conv blocks -----------------
    sim_gemm(XA, qnpack, S1, RN, wave, lane);
    conv_block_gemm(XA, XB, wpack1, b1, wave, lane);
    __syncthreads();
    conv_block_gemm(XB, XA, wpack2, b2, wave, lane);   // XA := sent_ctx
    __syncthreads();

    // ---- norms of sent_ctx --------------------------------------------------
    {
        int r = t >> 1, p = t & 1;
        const int pr = phys_row(r);
        float ss = 0.0f;
        for (int ch = p; ch < 75; ch += 2) {
#pragma unroll
            for (int j = 0; j < 4; ++j) {
                float xv = (float)XA[(size_t)pr * DP + ch * 4 + j];
                ss += xv * xv;
            }
        }
        PART[t] = ss;
    }
    __syncthreads();
    if (t < MROWS) RN[t] = rsqrtf(PART[2 * t] + PART[2 * t + 1] + 1e-20f);
    __syncthreads();

    sim_gemm(XA, qcnpack, S2, RN, wave, lane);
    __syncthreads();

    // ---- pooling (max / top5-mean / mean; oh-count) + MLP ------------------
    if (t < TS * NQ) {
        const int sl = t >> 5, q = t & 31;
        float t5i[5], t5s[5];
#pragma unroll
        for (int j = 0; j < 5; ++j) { t5i[j] = -1e30f; t5s[j] = -1e30f; }
        float sumi = 0.0f, sums = 0.0f; int cnt = 0;
        for (int l = 0; l < SL; ++l) {
            float vi = S1[(sl * SL + l) * NQ + q];
            sumi += vi;
            cnt += (vi > EPS_OH) ? 1 : 0;
            if (vi > t5i[0]) {
                t5i[0] = vi;
#pragma unroll
                for (int j = 0; j < 4; ++j)
                    if (t5i[j] > t5i[j + 1]) { float tm = t5i[j]; t5i[j] = t5i[j + 1]; t5i[j + 1] = tm; }
            }
            float vs = S2[(sl * SL + l) * NQ + q];
            sums += vs;
            if (vs > t5s[0]) {
                t5s[0] = vs;
#pragma unroll
                for (int j = 0; j < 4; ++j)
                    if (t5s[j] > t5s[j + 1]) { float tm = t5s[j]; t5s[j] = t5s[j + 1]; t5s[j + 1] = tm; }
            }
        }
        float f[9];
        f[0] = (cnt > 0) ? 1.0f : 0.0f;
        f[1] = (float)((cnt < 5) ? cnt : 5) * 0.2f;
        f[2] = (float)cnt * (1.0f / 64.0f);
        f[3] = t5i[4];
        f[4] = (t5i[0] + t5i[1] + t5i[2] + t5i[3] + t5i[4]) * 0.2f;
        f[5] = sumi * (1.0f / 64.0f);
        f[6] = t5s[4];
        f[7] = (t5s[0] + t5s[1] + t5s[2] + t5s[3] + t5s[4]) * 0.2f;
        f[8] = sums * (1.0f / 64.0f);

        float lo = m2b[0];
#pragma unroll
        for (int j = 0; j < 8; ++j) {
            float h = m1b[j];
#pragma unroll
            for (int i = 0; i < 9; ++i) h += f[i] * m1w[j * 9 + i];
            h = (h > 0.0f) ? h : 0.1f * h;
            lo += h * m2w[j];
        }
        LO[t] = lo * QWS[q];
    }
    __syncthreads();

    if (t < TS) {
        float e = 0.0f;
        for (int q = 0; q < NQ; ++q) e += LO[t * NQ + q];
        e *= (1.0f / 32.0f);
        const int s = blk * TS + t;
        float z = gaf[s * 3 + 0] * outw[0] + gaf[s * 3 + 1] * outw[1] +
                  gaf[s * 3 + 2] * outw[2] + e * outw[3];
        out[s] = 1.0f / (1.0f + expf(-z));
    }
}

// ---------------------------------------------------------------------------
extern "C" void kernel_launch(void* const* d_in, const int* in_sizes, int n_in,
                              void* d_out, int out_size, void* d_ws, size_t ws_size,
                              hipStream_t stream) {
    const float* doc  = (const float*)d_in[0];
    const float* qemb = (const float*)d_in[1];
    const float* qidf = (const float*)d_in[2];
    const float* gaf  = (const float*)d_in[3];
    const float* w1   = (const float*)d_in[4];
    const float* b1   = (const float*)d_in[5];
    const float* w2   = (const float*)d_in[6];
    const float* b2   = (const float*)d_in[7];
    const float* qww  = (const float*)d_in[8];
    const float* qwb  = (const float*)d_in[9];
    const float* m1w  = (const float*)d_in[10];
    const float* m1b  = (const float*)d_in[11];
    const float* m2w  = (const float*)d_in[12];
    const float* m2b  = (const float*)d_in[13];
    const float* outw = (const float*)d_in[14];

    char* ws = (char*)d_ws;
    bf16_t* wpack = (bf16_t*)(ws + WS_WPACK);
    bf16_t* qnp   = (bf16_t*)(ws + WS_QN);
    bf16_t* qcnp  = (bf16_t*)(ws + WS_QCN);
    float*  qwv   = (float*)(ws + WS_QW);
    float*  q1    = (float*)(ws + WS_Q1);
    float*  qctx  = (float*)(ws + WS_QCTX);
    float*  out   = (float*)d_out;

    // prep: weight pack (parallel) + question pipeline (tiny)
    pack_w_kernel<<<150, 256, 0, stream>>>(w1, w2, wpack);
    qconv_kernel<<<(NQ * DR + 255) / 256, 256, 0, stream>>>(qemb, w1, b1, q1);
    qconv_kernel<<<(NQ * DR + 255) / 256, 256, 0, stream>>>(q1, w2, b2, qctx);
    qfinal_kernel<<<1, 256, 0, stream>>>(qemb, qctx, qidf, qww, qwb, qnp, qcnp, qwv);

    // main: 2 sentences / block, 8 waves, dynamic LDS (~204KB)
    const size_t smem = (size_t)PROWS * DP * 2 * 2   // XA + XB (bf16, guard rows)
                      + (size_t)MROWS * NQ * 4 * 2   // S1 + S2
                      + MROWS * 4                    // RN
                      + 256 * 4                      // PART
                      + 64 * 4                       // LO
                      + NQ * 4;                      // QWS
    sent_main_kernel<<<4096 / TS, 256, smem, stream>>>(
        doc, gaf,
        wpack, wpack + (size_t)(3 * KT * NT_CONV) * 512,
        b1, b2, qnp, qcnp, qwv, m1w, m1b, m2w, m2b, outw, out);
}
